// Attention_43550968382196
// MI455X (gfx1250) — matrix operand order repaired and checked by reading, not compiled, not run on hardware
//
#include <hip/hip_runtime.h>
#include <hip/hip_bf16.h>

typedef __attribute__((ext_vector_type(16))) _Float16 v16h;
typedef __attribute__((ext_vector_type(8)))  float    v8f;

#define WMMA_F16(a, b, c) \
    __builtin_amdgcn_wmma_f32_16x16x32_f16(false, (a), false, (b), (short)0, (c), false, false)

#define MAXF(a, b) ((a) > (b) ? (a) : (b))

constexpr int   N_ROWS    = 32768;
constexpr int   NUM_BATCH = 64;
constexpr int   N_PER     = 512;          // rows per segment
constexpr int   DQK       = 128;
constexpr int   M_FEAT    = 256;
constexpr int   DV        = 256;
constexpr float INV_D4    = 0.29730177875068026f;  // 1/128^0.25
constexpr float H_SCALE   = 0.04419417382415922f;  // 1/(2*sqrt(128))
constexpr float INV_SQRTM = 0.0625f;               // 1/sqrt(256)
constexpr float EPS_PHI   = 1e-4f;
constexpr float EPS_NORM  = 1e-8f;

// Async copy: global -> LDS, 16 bytes per active lane, tracked by ASYNCcnt.
// Generic shared-pointer low 32 bits == LDS byte offset (ISA §10.2 aperture).
__device__ inline void async_copy_b128(void* lds_ptr, const void* gptr) {
    unsigned lofs = (unsigned)(unsigned long long)lds_ptr;
    unsigned long long g = (unsigned long long)gptr;
    asm volatile("global_load_async_to_lds_b128 %0, %1, off" :: "v"(lofs), "v"(g) : "memory");
}
__device__ inline void async_wait0() {
    asm volatile("s_wait_asynccnt 0x0" ::: "memory");
}

// ---------------------------------------------------------------------------
// omega [128][256] f32  ->  omegaT [256][128] f16  (so WMMA B-frag loads are
// contiguous 32B per lane)
// ---------------------------------------------------------------------------
__global__ __launch_bounds__(256) void omega_t_kernel(const float* __restrict__ omega,
                                                      _Float16* __restrict__ omegaT) {
    int idx = blockIdx.x * 256 + threadIdx.x;       // 128*256 elements
    int k = idx >> 8, n = idx & 255;
    omegaT[n * DQK + k] = (_Float16)omega[k * M_FEAT + n];
}

// ---------------------------------------------------------------------------
// Shared WMMA helper: 16x256 tile of U = (X * d^-1/4) @ omega into LDS.
// One wave per 16-row tile; 4 A-frags (K=128) reused over 16 N-tiles; 64 wmma.
// ---------------------------------------------------------------------------
__device__ inline void compute_u_tile(const float* __restrict__ X,
                                      const _Float16* __restrict__ omegaT,
                                      int row0, int l, float* uTile) {
    const int hi   = (l & 16) ? 1 : 0;
    const int koff = hi ? 8 : 0;     // unused; A-frag uses boff (contiguous K)
    const int boff = hi ? 16 : 0;    // B-frag K offset for lanes 16-31
    const int rl   = l & 15;

    const float* xrow = X + (size_t)(row0 + rl) * DQK;
    v16h afrag[4];
#pragma unroll
    for (int f = 0; f < 4; ++f) {
#pragma unroll
        for (int i = 0; i < 16; ++i) {
            int k = 32 * f + i + boff;   // contiguous K order: k = 16*hi + i (matches B)
            afrag[f][i] = (_Float16)(xrow[k] * INV_D4);
        }
    }
#pragma unroll 1
    for (int nt = 0; nt < 16; ++nt) {
        v8f c{};
        const _Float16* wcol = omegaT + (size_t)(nt * 16 + rl) * DQK;
#pragma unroll
        for (int f = 0; f < 4; ++f) {
            v16h b;
#pragma unroll
            for (int i = 0; i < 16; ++i) b[i] = wcol[32 * f + boff + i];
            c = WMMA_F16(afrag[f], b, c);
        }
#pragma unroll
        for (int rr = 0; rr < 8; ++rr)                        // C: row=rr+8*hi, col=rl
            uTile[(rr + 8 * hi) * M_FEAT + nt * 16 + rl] = c[rr];
    }
}

// All-32-lane helpers: 2 lanes per row (lane l handles row l&15, half l>>4).
__device__ inline void tile_rowmax(const float* uTile, int l, float* partial,
                                   float* mxS /*16*/) {
    const int r = l & 15, h0 = (l >> 4) * 128;
    float mx = -1e30f;
    for (int j = 0; j < 128; ++j) { float u = uTile[r * M_FEAT + h0 + j]; mx = MAXF(mx, u); }
    partial[l] = mx;
    __syncthreads();
    if (l < 16) mxS[l] = MAXF(partial[l], partial[l + 16]);
}

__device__ inline void tile_h(const float* __restrict__ X, int row0, int l,
                              float* partial, float* hS /*16*/) {
    const int r = l & 15;
    const float* xr = X + (size_t)(row0 + r) * DQK + (l >> 4) * 64;
    float s = 0.f;
    for (int k = 0; k < 64; ++k) { float v = xr[k]; s += v * v; }
    partial[l] = s;
    __syncthreads();
    if (l < 16) hS[l] = (partial[l] + partial[l + 16]) * H_SCALE;
}

// ---------------------------------------------------------------------------
// Qp = (exp(U - h - rowmax(U)) + eps) / sqrt(m), f16 row-major
// ---------------------------------------------------------------------------
__global__ __launch_bounds__(32) void phi_q_kernel(const float* __restrict__ Q,
                                                   const _Float16* __restrict__ omegaT,
                                                   _Float16* __restrict__ Qp) {
    __shared__ float uTile[16 * M_FEAT];
    __shared__ float partial[32];
    __shared__ float mxS[16], hS[16];
    const int row0 = blockIdx.x * 16;
    const int l = threadIdx.x;
    compute_u_tile(Q, omegaT, row0, l, uTile);
    __syncthreads();
    tile_rowmax(uTile, l, partial, mxS);
    __syncthreads();
    tile_h(Q, row0, l, partial, hS);
    __syncthreads();
    for (int e = l; e < 16 * M_FEAT; e += 32) {
        int r = e >> 8;
        float v = (__expf(uTile[e] - hS[r] - mxS[r]) + EPS_PHI) * INV_SQRTM;
        Qp[(size_t)(row0 + r) * M_FEAT + (e & 255)] = (_Float16)v;
    }
}

// ---------------------------------------------------------------------------
// Per-row max of U_K (phase 1 of the segment max)
// ---------------------------------------------------------------------------
__global__ __launch_bounds__(32) void k_rowmax_kernel(const float* __restrict__ K,
                                                      const _Float16* __restrict__ omegaT,
                                                      float* __restrict__ rowmax) {
    __shared__ float uTile[16 * M_FEAT];
    __shared__ float partial[32];
    __shared__ float mxS[16];
    const int row0 = blockIdx.x * 16;
    const int l = threadIdx.x;
    compute_u_tile(K, omegaT, row0, l, uTile);
    __syncthreads();
    tile_rowmax(uTile, l, partial, mxS);
    __syncthreads();
    if (l < 16) rowmax[row0 + l] = mxS[l];
}

__global__ __launch_bounds__(256) void segmax_kernel(const float* __restrict__ rowmax,
                                                     float* __restrict__ segmax) {
    __shared__ float sm[256];
    const int b = blockIdx.x, t = threadIdx.x;
    sm[t] = MAXF(rowmax[b * N_PER + t], rowmax[b * N_PER + 256 + t]);
    __syncthreads();
    for (int s = 128; s > 0; s >>= 1) {
        if (t < s) sm[t] = MAXF(sm[t], sm[t + s]);
        __syncthreads();
    }
    if (t == 0) segmax[b] = sm[0];
}

// ---------------------------------------------------------------------------
// Kp: recompute U_K by WMMA (cheaper than 32MB of U traffic at 23.3TB/s), then
// apply exp with per-segment max.
// ---------------------------------------------------------------------------
__global__ __launch_bounds__(32) void phi_k_kernel(const float* __restrict__ K,
                                                   const _Float16* __restrict__ omegaT,
                                                   const float* __restrict__ segmax,
                                                   _Float16* __restrict__ Kp) {
    __shared__ float uTile[16 * M_FEAT];
    __shared__ float partial[32];
    __shared__ float hS[16];
    const int row0 = blockIdx.x * 16;
    const int l = threadIdx.x;
    compute_u_tile(K, omegaT, row0, l, uTile);
    __syncthreads();
    tile_h(K, row0, l, partial, hS);
    __syncthreads();
    const float mx = segmax[row0 >> 9];     // 512 rows per segment
    for (int e = l; e < 16 * M_FEAT; e += 32) {
        int r = e >> 8;
        float v = (__expf(uTile[e] - hS[r] - mx) + EPS_PHI) * INV_SQRTM;
        Kp[(size_t)(row0 + r) * M_FEAT + (e & 255)] = (_Float16)v;
    }
}

// ---------------------------------------------------------------------------
// Ksum[b][m] = sum_j Kp[b*512+j][m]   (coalesced column sums + prefetch)
// ---------------------------------------------------------------------------
__global__ __launch_bounds__(256) void ksum_kernel(const _Float16* __restrict__ Kp,
                                                   float* __restrict__ Ksum) {
    const int b = blockIdx.x, m = threadIdx.x;
    const _Float16* col = Kp + (size_t)b * N_PER * M_FEAT + m;
    float s = 0.f;
    for (int j = 0; j < N_PER; ++j) {
        if ((j & 31) == 0 && j + 32 < N_PER)
            __builtin_prefetch(col + (size_t)(j + 32) * M_FEAT, 0, 1);
        s += (float)col[(size_t)j * M_FEAT];
    }
    Ksum[b * M_FEAT + m] = s;
}

__global__ __launch_bounds__(256) void norm_kernel(const _Float16* __restrict__ Qp,
                                                   const float* __restrict__ Ksum,
                                                   float* __restrict__ normv) {
    const int i = blockIdx.x * 256 + threadIdx.x;
    const float* ks = Ksum + (size_t)(i >> 9) * M_FEAT;
    const _Float16* q = Qp + (size_t)i * M_FEAT;
    float s = 0.f;
    for (int m = 0; m < M_FEAT; ++m) s += (float)q[m] * ks[m];
    normv[i] = s + EPS_NORM;
}

// ---------------------------------------------------------------------------
// KV = Kp^T @ V per batch, stored transposed: KVT[b][vcol][feat] f16.
// Block = 256 threads (8 waves) handles (batch, 16-feature tile).
// Kp chunk staged row-major via GLOBAL_LOAD_ASYNC_TO_LDS_B128 (A-frag is a
// transposed LDS gather); V chunk staged manually (f32->f16 conversion) and
// transposed so B-frag reads are contiguous.  32 v_wmma per wave.
// ---------------------------------------------------------------------------
__global__ __launch_bounds__(256) void kv_kernel(const _Float16* __restrict__ Kp,
                                                 const float* __restrict__ V,
                                                 _Float16* __restrict__ KVT) {
    const int b = blockIdx.x >> 4, mt = blockIdx.x & 15;
    const int t = threadIdx.x, wave = t >> 5, l = t & 31;
    const int hi = (l & 16) ? 1 : 0, rl = l & 15;
    const int koff = hi ? 8 : 0, boff = hi ? 16 : 0;

    __shared__ __align__(16) _Float16 ldsKp[32 * 16];  // [row][feat_local] row-major
    __shared__ __align__(16) _Float16 ldsV[DV * 32];   // [vcol][row] transposed

    v8f acc0{}, acc1{};

    for (int k0 = 0; k0 < N_PER; k0 += 32) {
        // async-stage Kp chunk 32x16 f16 (1KB): rows stride 512B, 2x16B per row
        if (t < 64) {
            int row = t >> 1, half = (t & 1) * 8;
            async_copy_b128(&ldsKp[row * 16 + half],
                            Kp + (size_t)(b * N_PER + k0 + row) * M_FEAT + mt * 16 + half);
        }
        // stage V chunk 32x256, convert f32->f16, transpose into [vcol][row]
        for (int j = 0; j < 32; ++j)
            ldsV[t * 32 + j] = (_Float16)V[(size_t)(b * N_PER + k0 + j) * DV + t];
        async_wait0();
        __syncthreads();

        v16h a;                                            // A = Kp^T tile (LDS gather)
#pragma unroll
        for (int i = 0; i < 16; ++i) {
            int k = i + boff;
            a[i] = ldsKp[k * 16 + rl];
        }

        v16h b0, b1;                                       // B = V tile
        const _Float16* v0 = &ldsV[(wave * 32 + rl) * 32 + boff];
        const _Float16* v1 = &ldsV[(wave * 32 + 16 + rl) * 32 + boff];
#pragma unroll
        for (int i = 0; i < 16; ++i) { b0[i] = v0[i]; b1[i] = v1[i]; }

        acc0 = WMMA_F16(a, b0, acc0);
        acc1 = WMMA_F16(a, b1, acc1);
        __syncthreads();
    }

    // store: KVT[b][vcol][feat]; per-lane 8 contiguous f16 in feat
#pragma unroll
    for (int rr = 0; rr < 8; ++rr) {
        int feat = mt * 16 + rr + 8 * hi;
        KVT[((size_t)b * DV + wave * 32 + rl) * M_FEAT + feat]      = (_Float16)acc0[rr];
        KVT[((size_t)b * DV + wave * 32 + 16 + rl) * M_FEAT + feat] = (_Float16)acc1[rr];
    }
}

// ---------------------------------------------------------------------------
// out = (Qp @ KV) / norm per batch.  Qp tile async-staged to LDS; KVT B-frag
// reads are contiguous 32B per lane from global.
// ---------------------------------------------------------------------------
__global__ __launch_bounds__(256) void out_kernel(const _Float16* __restrict__ Qp,
                                                  const _Float16* __restrict__ KVT,
                                                  const float* __restrict__ normv,
                                                  float* __restrict__ out) {
    const int b = blockIdx.x >> 5, rt = blockIdx.x & 31;
    const int row0 = b * N_PER + rt * 16;
    const int t = threadIdx.x, wave = t >> 5, l = t & 31;
    const int hi = (l & 16) ? 1 : 0, rl = l & 15;
    const int koff = hi ? 8 : 0, boff = hi ? 16 : 0;

    __shared__ __align__(16) _Float16 ldsQ[16 * M_FEAT];   // 8KB
    __shared__ float invn[16];

    // async-stage the 16x256 Qp tile: 2x 16B per thread
    const _Float16* gsrc = Qp + (size_t)row0 * M_FEAT;
    async_copy_b128(&ldsQ[t * 8], gsrc + t * 8);
    async_copy_b128(&ldsQ[2048 + t * 8], gsrc + 2048 + t * 8);
    if (t < 16) invn[t] = 1.0f / normv[row0 + t];
    async_wait0();
    __syncthreads();

    v8f acc0{}, acc1{};
#pragma unroll 1
    for (int k0 = 0; k0 < M_FEAT; k0 += 32) {
        v16h a;
#pragma unroll
        for (int i = 0; i < 16; ++i)
            a[i] = ldsQ[rl * M_FEAT + k0 + i + boff];

        const _Float16* s0 = KVT + ((size_t)b * DV + wave * 32 + rl) * M_FEAT + k0 + boff;
        const _Float16* s1 = KVT + ((size_t)b * DV + wave * 32 + 16 + rl) * M_FEAT + k0 + boff;
        v16h b0, b1;
#pragma unroll
        for (int i = 0; i < 16; ++i) { b0[i] = s0[i]; b1[i] = s1[i]; }

        acc0 = WMMA_F16(a, b0, acc0);
        acc1 = WMMA_F16(a, b1, acc1);
    }

#pragma unroll
    for (int rr = 0; rr < 8; ++rr) {
        int r = rr + 8 * hi;
        float inv = invn[r];
        out[(size_t)(row0 + r) * DV + wave * 32 + rl]      = acc0[rr] * inv;
        out[(size_t)(row0 + r) * DV + wave * 32 + 16 + rl] = acc1[rr] * inv;
    }
}

// ---------------------------------------------------------------------------
extern "C" void kernel_launch(void* const* d_in, const int* in_sizes, int n_in,
                              void* d_out, int out_size, void* d_ws, size_t ws_size,
                              hipStream_t stream) {
    (void)in_sizes; (void)n_in; (void)out_size; (void)ws_size;
    const float* Q     = (const float*)d_in[0];
    const float* K     = (const float*)d_in[1];
    const float* V     = (const float*)d_in[2];
    const float* omega = (const float*)d_in[3];
    float* out = (float*)d_out;

    char* ws = (char*)d_ws;
    const size_t MB = 1024ull * 1024ull;
    _Float16* Qp     = (_Float16*)(ws);                 // 16 MB
    _Float16* Kp     = (_Float16*)(ws + 16 * MB);       // 16 MB
    _Float16* KVT    = (_Float16*)(ws + 32 * MB);       //  8 MB
    _Float16* omegaT = (_Float16*)(ws + 40 * MB);       // 64 KB
    float* rowmax    = (float*)(ws + 40 * MB + 256 * 1024);   // 128 KB
    float* segmax    = (float*)(ws + 40 * MB + 512 * 1024);   // 256 B
    float* Ksum      = (float*)(ws + 40 * MB + 640 * 1024);   // 64 KB
    float* normv     = (float*)(ws + 40 * MB + 768 * 1024);   // 128 KB

    omega_t_kernel<<<DQK, 256, 0, stream>>>(omega, omegaT);
    phi_q_kernel<<<N_ROWS / 16, 32, 0, stream>>>(Q, omegaT, Qp);
    k_rowmax_kernel<<<N_ROWS / 16, 32, 0, stream>>>(K, omegaT, rowmax);
    segmax_kernel<<<NUM_BATCH, 256, 0, stream>>>(rowmax, segmax);
    phi_k_kernel<<<N_ROWS / 16, 32, 0, stream>>>(K, omegaT, segmax, Kp);
    ksum_kernel<<<NUM_BATCH, 256, 0, stream>>>(Kp, Ksum);
    norm_kernel<<<N_ROWS / 256, 256, 0, stream>>>(Qp, Ksum, normv);
    kv_kernel<<<NUM_BATCH * 16, 256, 0, stream>>>(Kp, V, KVT);
    out_kernel<<<NUM_BATCH * 32, 256, 0, stream>>>(Qp, KVT, normv, out);
}
